// TopLineEncoder_8400956031557
// MI455X (gfx1250) — matrix operand order repaired and checked by reading, not compiled, not run on hardware
//
#include <hip/hip_runtime.h>
#include <hip/hip_bf16.h>

#define HID  128
#define NPOS 80

typedef __attribute__((ext_vector_type(16))) _Float16 v16h;
typedef __attribute__((ext_vector_type(8)))  _Float16 v8h;
typedef __attribute__((ext_vector_type(4)))  _Float16 v4h;
typedef __attribute__((ext_vector_type(8)))  float    v8f;

#define WLDS_BYTES  (3 * HID * HID * 2)          // 98304: packed f16 W2/W3/W4
#define HBUF_BYTES  (4 * 16 * HID * 2)           // 16384: per-wave activations
#define CBUF_BYTES  (4 * NPOS * 4)               // 1280 : per-wave char staging
#define SMEM_BYTES  (WLDS_BYTES + HBUF_BYTES + CBUF_BYTES)

__device__ __forceinline__ float elu_f(float x) {
    return x > 0.0f ? x : (__expf(x) - 1.0f);
}

// ---------------------------------------------------------------------------
// Pre-pack W2/W3/W4 (fp32, row-major [in=K][out=N]) into f16 fragment-major
// layout for v_wmma_f32_16x16x32_f16 B-operands (32x16 KxN per fragment):
// for lane (half = lane>>4, n = nt*16 + (lane&15)), element j holds
//     W[(kt*32 + half*16 + j) * HID + n]
// ---------------------------------------------------------------------------
__global__ void pack_weights_kernel(const float* __restrict__ W2,
                                    const float* __restrict__ W3,
                                    const float* __restrict__ W4,
                                    _Float16* __restrict__ Wp) {
    int t = blockIdx.x * blockDim.x + threadIdx.x;
    if (t >= 3 * 4 * 8 * 32) return;
    const int lane = t & 31;
    const int nt   = (t >> 5) & 7;
    const int kt   = (t >> 8) & 3;
    const int ly   = t >> 10;
    const float* W = (ly == 0) ? W2 : (ly == 1) ? W3 : W4;
    const int half = lane >> 4;
    const int n    = nt * 16 + (lane & 15);
    _Float16* dst = Wp + (size_t)((((ly * 4 + kt) * 8 + nt) * 32) + lane) * 16;
#pragma unroll
    for (int j = 0; j < 16; ++j) {
        const int k = kt * 32 + half * 16 + j;
        dst[j] = (_Float16)W[k * HID + n];
    }
}

// ---------------------------------------------------------------------------
// Fused 4-layer MLP. 128 threads = 4 waves; each wave owns 16 batch rows and
// the full 128-wide hidden dim (8 N-tiles of v8f WMMA accumulators).
// Packed weights are async-copied global->LDS (ASYNCcnt) overlapping the
// fp32 embedding gather, then all GEMM B-operands come from LDS.
// ---------------------------------------------------------------------------
__global__ __launch_bounds__(128) void mlp_fused_kernel(
    const int*   __restrict__ msg,   // [B][80] (int32 on device)
    const float* __restrict__ W1,    // [80*256][128] fp32
    const float* __restrict__ b1,
    const float* __restrict__ b2,
    const float* __restrict__ b3,
    const float* __restrict__ b4,
    const _Float16* __restrict__ Wp, // packed f16 fragments for W2/W3/W4
    float* __restrict__ out)         // [B][128] fp32
{
    extern __shared__ __align__(16) char smem_raw[];
    _Float16* wlds = (_Float16*)smem_raw;                                  // 96 KB
    const int lane = threadIdx.x & 31;
    const int wv   = threadIdx.x >> 5;
    const int half = lane >> 4;
    const int nn   = lane & 15;
    _Float16* hb   = (_Float16*)(smem_raw + WLDS_BYTES) + wv * (16 * HID); // 4 KB/wave
    int*      cbuf = (int*)(smem_raw + WLDS_BYTES + HBUF_BYTES) + wv * NPOS;

    const int row0 = (blockIdx.x * 4 + wv) * 16;

    // ---- Kick off async global->LDS copy of all packed weights (96 KB). ----
    // Tracked by ASYNCcnt; overlaps with the layer-1 gather below.
    {
        const uint32_t lbase = (uint32_t)(uintptr_t)smem_raw + threadIdx.x * 16u;
        const char*    gsrc  = (const char*)Wp + threadIdx.x * 16;
#pragma unroll 4
        for (int it = 0; it < WLDS_BYTES / 2048; ++it) {   // 48 iters * 2 KB
            const uint32_t loff = lbase + (uint32_t)(it * 2048);
            const void*    g    = gsrc + it * 2048;
            asm volatile("global_load_async_to_lds_b128 %0, %1, off"
                         :: "v"(loff), "v"(g) : "memory");
        }
    }

    // ---------------- Layer 1: exact fp32 embedding gather-sum ----------------
    // Each lane owns 4 consecutive hid columns (32 lanes * 4 = 128).
    const float4 bias1 = *(const float4*)(b1 + lane * 4);
    for (int r = 0; r < 16; ++r) {
        const int* mrow = msg + (size_t)(row0 + r) * NPOS;
        // Stage this row's 80 chars into LDS (3 loads/wave instead of 80
        // broadcast global loads in the inner loop).
        if (lane < 16) cbuf[64 + lane] = mrow[64 + lane];
        cbuf[lane]      = mrow[lane];
        cbuf[32 + lane] = mrow[32 + lane];

        float ax = 0.0f, ay = 0.0f, az = 0.0f, aw = 0.0f;
#pragma unroll 4
        for (int p = 0; p < NPOS; ++p) {
            const int c = cbuf[p];
            if (p + 8 < NPOS) {   // prefetch data-dependent W1 row ahead
                __builtin_prefetch(
                    W1 + (size_t)((p + 8) * 256 + cbuf[p + 8]) * HID + lane * 4,
                    0, 3);
            }
            const float4 w =
                *(const float4*)(W1 + (size_t)(p * 256 + c) * HID + lane * 4);
            ax += w.x; ay += w.y; az += w.z; aw += w.w;
        }
        v4h hv;
        hv[0] = (_Float16)elu_f(ax + bias1.x);
        hv[1] = (_Float16)elu_f(ay + bias1.y);
        hv[2] = (_Float16)elu_f(az + bias1.z);
        hv[3] = (_Float16)elu_f(aw + bias1.w);
        *(v4h*)&hb[r * HID + lane * 4] = hv;
    }

    // Weights must be resident in LDS before the GEMM phase.
    asm volatile("s_wait_asynccnt 0x0" ::: "memory");
    __syncthreads();

    // ---------------- Layers 2..4: WMMA f16->f32 GEMM + bias + ELU -----------
    for (int ly = 0; ly < 3; ++ly) {
        v8f acc[8];
        const v8f vzero = {0.f, 0.f, 0.f, 0.f, 0.f, 0.f, 0.f, 0.f};
#pragma unroll
        for (int nt = 0; nt < 8; ++nt) acc[nt] = vzero;

#pragma unroll
        for (int kt = 0; kt < 4; ++kt) {
            // A fragment: this code's contiguous k order, same as the packed B.
            // lane row m = nn; element j: K = 16*half + j.
            const v8h lo = *(const v8h*)&hb[nn * HID + kt * 32 + half * 16];
            const v8h hi = *(const v8h*)&hb[nn * HID + kt * 32 + half * 16 + 8];
            v16h a;
#pragma unroll
            for (int i = 0; i < 8; ++i) { a[i] = lo[i]; a[i + 8] = hi[i]; }

            // Batch-load all 8 B fragments, then fence the scheduler so the
            // ds_load_b128s cannot be sunk down next to their WMMA consumers.
            // DScnt decrements in order, so the WMMAs can then retire against
            // staged s_wait_dscnt counts instead of wait(0) per tile.
            const _Float16* wbase = wlds + (size_t)((ly * 4 + kt) * 8) * 32 * 16;
            v16h bfrag[8];
#pragma unroll
            for (int nt = 0; nt < 8; ++nt) {
                bfrag[nt] = *(const v16h*)(wbase + (size_t)(nt * 32 + lane) * 16);
            }
            __builtin_amdgcn_sched_barrier(0);   // no reordering across this point
#pragma unroll
            for (int nt = 0; nt < 8; ++nt) {
                acc[nt] = __builtin_amdgcn_wmma_f32_16x16x32_f16(
                    false, a, false, bfrag[nt], (short)0, acc[nt], false, false);
            }
        }

        const float* bl = (ly == 0) ? b2 : (ly == 1) ? b3 : b4;
        __syncthreads();   // all A-fragment reads done before hb overwrite

        if (ly < 2) {
            // D-layout: VGPR v, lane -> (M = 8*half + v, N = nt*16 + nn)
#pragma unroll
            for (int nt = 0; nt < 8; ++nt) {
                const float bv = bl[nt * 16 + nn];
#pragma unroll
                for (int v = 0; v < 8; ++v) {
                    const float x = elu_f(acc[nt][v] + bv);
                    hb[(half * 8 + v) * HID + nt * 16 + nn] = (_Float16)x;
                }
            }
            __syncthreads();
        } else {
            float* orow = out + (size_t)row0 * HID;
#pragma unroll
            for (int nt = 0; nt < 8; ++nt) {
                const float bv = bl[nt * 16 + nn];
#pragma unroll
                for (int v = 0; v < 8; ++v) {
                    orow[(size_t)(half * 8 + v) * HID + nt * 16 + nn] =
                        elu_f(acc[nt][v] + bv);
                }
            }
        }
    }
}

extern "C" void kernel_launch(void* const* d_in, const int* in_sizes, int n_in,
                              void* d_out, int out_size, void* d_ws, size_t ws_size,
                              hipStream_t stream) {
    const int*   msg = (const int*)  d_in[0];
    const float* W1  = (const float*)d_in[1];
    const float* b1  = (const float*)d_in[2];
    const float* W2  = (const float*)d_in[3];
    const float* b2  = (const float*)d_in[4];
    const float* W3  = (const float*)d_in[5];
    const float* b3  = (const float*)d_in[6];
    const float* W4  = (const float*)d_in[7];
    const float* b4  = (const float*)d_in[8];
    float*    out = (float*)d_out;
    _Float16* Wp  = (_Float16*)d_ws;   // 3*128*128 f16 = 96 KB packed weights

    const int B = in_sizes[0] / NPOS;  // 8192

    pack_weights_kernel<<<(3 * 4 * 8 * 32 + 127) / 128, 128, 0, stream>>>(
        W2, W3, W4, Wp);
    mlp_fused_kernel<<<B / 64, 128, SMEM_BYTES, stream>>>(
        msg, W1, b1, b2, b3, b4, Wp, out);
}
